// ModernBertAttention_16973710753966
// MI455X (gfx1250) — matrix operand order repaired and checked by reading, not compiled, not run on hardware
//
#include <hip/hip_runtime.h>
#include <hip/hip_bf16.h>
#include <math.h>
#include <cstdint>

// ---- problem constants (match reference) ----
#define BB   2
#define SS   2048
#define HID_ 768
#define NH_  12
#define HD_  64
#define ROPE_THETA 10000.0f

typedef __attribute__((ext_vector_type(16))) __bf16 v16bf;
typedef __attribute__((ext_vector_type(8)))  float  v8f;

// ---- CDNA5 async global->LDS path (guarded; fallback = ds stores) ----
#if defined(__has_builtin)
#  if __has_builtin(__builtin_amdgcn_global_load_async_to_lds_b128)
#    define HAVE_ASYNC_LDS 1
#  endif
#  if __has_builtin(__builtin_amdgcn_s_wait_asynccnt)
#    define HAVE_WAIT_ASYNC 1
#  endif
#endif

// exact parameter types per hipcc diagnostic: pointer to vector_size(16) int,
// AS1 (global) source / AS3 (LDS) destination
typedef int v4i_gcc __attribute__((vector_size(16)));
typedef v4i_gcc __attribute__((address_space(1)))* g_v4i_ptr;
typedef v4i_gcc __attribute__((address_space(3)))* l_v4i_ptr;

__device__ __forceinline__ void async_copy16(const float* gsrc, float* lds_dst) {
#ifdef HAVE_ASYNC_LDS
    g_v4i_ptr gp = (g_v4i_ptr)(uintptr_t)gsrc;
    l_v4i_ptr lp = (l_v4i_ptr)(uintptr_t)lds_dst;   // low 32 bits = LDS offset
    __builtin_amdgcn_global_load_async_to_lds_b128(gp, lp, 0, 0);
#else
    *(float4*)lds_dst = *(const float4*)gsrc;
#endif
}

__device__ __forceinline__ void wait_async_lds() {
#ifdef HAVE_ASYNC_LDS
#  ifdef HAVE_WAIT_ASYNC
    __builtin_amdgcn_s_wait_asynccnt(0);
#  else
    asm volatile("s_wait_asynccnt 0" ::: "memory");
#  endif
#endif
}

// =====================================================================
// Kernel 1/4: C[M,N] = A[M,K] * W[N,K]^T   (fp32 in/out, bf16 WMMA math)
// 128 threads = 4 waves; block tile 64x64; wave tile 32x32 (2x2 WMMA).
// Tiles staged fp32 via async global->LDS DMA; cvt to bf16 at frag build.
// =====================================================================
#define TM 64
#define TN 64
#define TK 32

__global__ __launch_bounds__(128)
void gemm_bf16wmma(const float* __restrict__ A, const float* __restrict__ W,
                   float* __restrict__ C, int M, int N, int K)
{
    __shared__ float sAf[TM][TK];   // 8 KB (fp32 tile, async-staged)
    __shared__ float sWf[TN][TK];   // 8 KB

    const int tid  = threadIdx.x;
    const int wave = tid >> 5;
    const int lane = tid & 31;
    const int bm = blockIdx.x * TM;
    const int bn = blockIdx.y * TN;
    const int m0 = (wave >> 1) * 32;        // wave quadrant
    const int n0 = (wave & 1)  * 32;

    v8f acc[2][2] = {};

    for (int k0 = 0; k0 < K; k0 += TK) {
        // ---- stage fp32 tiles: 64x32 = 512 x 16B chunks, 4 per thread ----
        for (int i = tid; i < (TM * TK) / 4; i += 128) {
            int r = i >> 3, cc = (i & 7) * 4;
            async_copy16(&A[(size_t)(bm + r) * K + k0 + cc], &sAf[r][cc]);
        }
        for (int i = tid; i < (TN * TK) / 4; i += 128) {
            int r = i >> 3, cc = (i & 7) * 4;
            async_copy16(&W[(size_t)(bn + r) * K + k0 + cc], &sWf[r][cc]);
        }
        // prefetch next K tiles into cache while this step computes
        if (k0 + TK < K) {
            __builtin_prefetch(&A[(size_t)(bm + (tid >> 1)) * K + k0 + TK], 0, 1);
            __builtin_prefetch(&W[(size_t)(bn + (tid >> 1)) * K + k0 + TK], 0, 1);
        }
        wait_async_lds();
        __syncthreads();

        const int rowm = lane & 15;
        const int kA   = (lane < 16) ? 0 : 16;   // A frag: K=0-15 vs 16-31
        const int kB   = (lane < 16) ? 0 : 16;   // B frag: K=0-15 vs 16-31

        v16bf afr[2], bfr[2];
        for (int mi = 0; mi < 2; ++mi) {
            int r = m0 + 16 * mi + rowm;
            for (int j = 0; j < 8; ++j) {
                afr[mi][j]     = (__bf16)sAf[r][kA + j];
                afr[mi][8 + j] = (__bf16)sAf[r][kA + 8 + j];
            }
        }
        for (int ni = 0; ni < 2; ++ni) {
            int r = n0 + 16 * ni + rowm;
            for (int j = 0; j < 16; ++j)
                bfr[ni][j] = (__bf16)sWf[r][kB + j];
        }
        for (int mi = 0; mi < 2; ++mi)
            for (int ni = 0; ni < 2; ++ni)
                acc[mi][ni] = __builtin_amdgcn_wmma_f32_16x16x32_bf16(
                    false, afr[mi], false, bfr[ni], (short)0, acc[mi][ni], false, false);
        __syncthreads();
    }

    // C layout: lane l, vgpr j -> M = j + 8*(l>=16), N = l%16
    const int ncol = lane & 15;
    const int moff = (lane >> 4) * 8;
    for (int mi = 0; mi < 2; ++mi)
        for (int ni = 0; ni < 2; ++ni) {
            int cbase = bn + n0 + 16 * ni + ncol;
            for (int j = 0; j < 8; ++j) {
                int r = bm + m0 + 16 * mi + moff + j;
                C[(size_t)r * N + cbase] = acc[mi][ni][j];
            }
        }
}

// =====================================================================
// Kernel 2/4: RoPE.  Qrot = rope(Q).  One thread per (b,s,h,freq) pair.
// =====================================================================
__global__ void rope_kernel(const float* __restrict__ Q,
                            const int* __restrict__ pos_idx,
                            float* __restrict__ Qrot)
{
    int idx = blockIdx.x * blockDim.x + threadIdx.x;
    const int total = BB * SS * NH_ * 32;
    if (idx >= total) return;
    int i   = idx & 31;               // frequency index 0..31
    int t   = idx >> 5;
    int h   = t % NH_;
    int s   = (t / NH_) % SS;
    int b   = t / (NH_ * SS);

    float pos = (float)pos_idx[s];
    float inv_freq = __powf(ROPE_THETA, -(2.0f * (float)i) / (float)HD_);
    float f = pos * inv_freq;
    float sn, cs;
    __sincosf(f, &sn, &cs);

    size_t base = ((size_t)b * SS + s) * HID_ + h * HD_;
    float x1 = Q[base + i];
    float x2 = Q[base + 32 + i];
    Qrot[base + i]      = x1 * cs - x2 * sn;   // first half:  x*cos - rot*sin
    Qrot[base + 32 + i] = x2 * cs + x1 * sn;   // second half: x*cos + rot*sin
}

// =====================================================================
// Kernel 3/4: sliding-window attention, 1 wave per (16-query tile, h, b).
// Keys staged: 10 tiles of 16 covering [t0-64, t0+95]; masked to |d|<=64.
// =====================================================================
#define NKEY 160
#define SCALE 0.125f   // HD^-0.5

__global__ __launch_bounds__(32)
void attn_kernel(const float* __restrict__ Qrot,   // queries & keys (rope'd)
                 const float* __restrict__ V,      // values (pre-rope Q)
                 float* __restrict__ O)
{
    __shared__ float  sScore[16][NKEY];   // 10 KB
    __shared__ __bf16 sP[16][NKEY];       // 5 KB
    __shared__ __bf16 sV[NKEY][HD_];      // 20 KB

    const int lane = threadIdx.x;
    const int qt = blockIdx.x, h = blockIdx.y, b = blockIdx.z;
    const int t0   = qt * 16;
    const int key0 = t0 - 64;             // 16-aligned window start

    const size_t headoff = (size_t)h * HD_;
    const size_t rowstride = HID_;
    const size_t bbase = (size_t)b * SS * HID_;

    // ---- load Q A-fragments (2 K-chunks of 32 over HD=64), fp32->bf16 ----
    const int rowm = lane & 15;
    const int kA   = (lane < 16) ? 0 : 16;
    const int kB   = (lane < 16) ? 0 : 16;
    v16bf aq[2];
    {
        const float* qrow = Qrot + bbase + (size_t)(t0 + rowm) * rowstride + headoff;
        for (int c = 0; c < 2; ++c)
            for (int j = 0; j < 8; ++j) {
                aq[c][j]     = (__bf16)qrow[32 * c + kA + j];
                aq[c][8 + j] = (__bf16)qrow[32 * c + kA + 8 + j];
            }
    }

    // ---- stage V tile (160 x 64) to LDS as bf16, clamped rows ----
    for (int i = lane; i < NKEY * HD_; i += 32) {
        int r = i >> 6, c = i & 63;
        int key = key0 + r;
        int kc  = key < 0 ? 0 : (key >= SS ? SS - 1 : key);
        sV[r][c] = (__bf16)V[bbase + (size_t)kc * rowstride + headoff + c];
    }

    // ---- scores: for each of 10 key tiles, 2 chained WMMAs (K=64) ----
    for (int kt = 0; kt < 10; ++kt) {
        int key = key0 + kt * 16 + rowm;      // this lane's key column
        int kc  = key < 0 ? 0 : (key >= SS ? SS - 1 : key);
        const float* krow = Qrot + bbase + (size_t)kc * rowstride + headoff;
        v16bf bk0, bk1;
        for (int j = 0; j < 16; ++j) {
            bk0[j] = (__bf16)krow[kB + j];          // K-dim 0..31
            bk1[j] = (__bf16)krow[32 + kB + j];     // K-dim 32..63
        }
        v8f sc = {};
        sc = __builtin_amdgcn_wmma_f32_16x16x32_bf16(false, aq[0], false, bk0, (short)0, sc, false, false);
        sc = __builtin_amdgcn_wmma_f32_16x16x32_bf16(false, aq[1], false, bk1, (short)0, sc, false, false);
        // scatter C fragment into LDS score tile (scaled)
        int moff = (lane >> 4) * 8;
        for (int j = 0; j < 8; ++j)
            sScore[moff + j][kt * 16 + rowm] = sc[j] * SCALE;
    }
    __syncthreads();

    // ---- masked softmax: lanes 0..15 each own one query row ----
    if (lane < 16) {
        int q = t0 + lane;
        float mx = -3.0e38f;
        for (int c = 0; c < NKEY; ++c) {
            int key = key0 + c;
            bool valid = (key >= 0) && (key < SS) && (abs(key - q) <= 64);
            float s = valid ? sScore[lane][c] : -3.0e38f;
            sScore[lane][c] = s;
            mx = fmaxf(mx, s);
        }
        float sum = 0.0f;
        for (int c = 0; c < NKEY; ++c) {
            float s = sScore[lane][c];
            float p = (s > -1.0e38f) ? __expf(s - mx) : 0.0f;
            sScore[lane][c] = p;
            sum += p;
        }
        float inv = 1.0f / sum;
        for (int c = 0; c < NKEY; ++c)
            sP[lane][c] = (__bf16)(sScore[lane][c] * inv);
    }
    __syncthreads();

    // ---- O = P(16x160) @ V(160x64): 5 K-chunks x 4 N-chunks of WMMA ----
    v8f oacc[4] = {};
    for (int kcn = 0; kcn < 5; ++kcn) {
        v16bf ap;
        for (int j = 0; j < 8; ++j) {
            ap[j]     = sP[rowm][32 * kcn + kA + j];
            ap[8 + j] = sP[rowm][32 * kcn + kA + 8 + j];
        }
        for (int nc = 0; nc < 4; ++nc) {
            v16bf bv;
            int d = 16 * nc + rowm;
            for (int j = 0; j < 16; ++j)
                bv[j] = sV[32 * kcn + kB + j][d];
            oacc[nc] = __builtin_amdgcn_wmma_f32_16x16x32_bf16(
                false, ap, false, bv, (short)0, oacc[nc], false, false);
        }
    }

    // ---- write O (fp32) ----
    {
        int moff = (lane >> 4) * 8;
        for (int nc = 0; nc < 4; ++nc)
            for (int j = 0; j < 8; ++j)
                O[bbase + (size_t)(t0 + moff + j) * rowstride + headoff + 16 * nc + rowm]
                    = oacc[nc][j];
    }
}

// =====================================================================
// launch
// =====================================================================
extern "C" void kernel_launch(void* const* d_in, const int* in_sizes, int n_in,
                              void* d_out, int out_size, void* d_ws, size_t ws_size,
                              hipStream_t stream) {
    const float* hidden = (const float*)d_in[0];   // [B,S,HID]
    const float* Wq     = (const float*)d_in[1];   // [HID,HID]
    const float* Wo     = (const float*)d_in[2];   // [HID,HID]
    // d_in[3] local mask, d_in[4] global mask: encoded analytically in-kernel
    const int*   pos    = (const int*)d_in[5];     // [1,S]
    float* out = (float*)d_out;

    const size_t nQ = (size_t)BB * SS * HID_;
    float* Q    = (float*)d_ws;     // pre-rope projection (= V)
    float* Qrot = Q + nQ;           // rope'd (= queries and keys)
    float* AO   = Qrot + nQ;        // attention output

    const int M = BB * SS;          // 4096
    dim3 gemmGrid(M / TM, HID_ / TN);

    gemm_bf16wmma<<<gemmGrid, 128, 0, stream>>>(hidden, Wq, Q, M, HID_, HID_);

    int ropeTotal = BB * SS * NH_ * 32;
    rope_kernel<<<(ropeTotal + 255) / 256, 256, 0, stream>>>(Q, pos, Qrot);

    attn_kernel<<<dim3(SS / 16, NH_, BB), 32, 0, stream>>>(Qrot, Q, AO);

    gemm_bf16wmma<<<gemmGrid, 128, 0, stream>>>(AO, Wo, out, M, HID_, HID_);
}
